// BrainDynaMo_85985245266497
// MI455X (gfx1250) — matrix operand order repaired and checked by reading, not compiled, not run on hardware
//
#include <hip/hip_runtime.h>

#define BB   16
#define TT   150
#define CC   100
#define EE   64
#define HH   128
#define CPAD 112      // C padded to multiple of 16
#define MT7  7        // CPAD/16

// classifier geometry
#define M_CLF   2400          // B*T rows
#define M_PAD   2432          // padded rows for guard-free async reads
#define K1      10000         // C*C
#define K1P     10016         // K1 padded to 32
#define N1      5000
#define N1P     5120          // padded to 128
#define K2P     5024          // 5000 padded to 32
#define N2      2500
#define N2P     2560

typedef __attribute__((ext_vector_type(8)))  float  v8f;
typedef __attribute__((ext_vector_type(16))) __bf16 v16bf;

struct Frag { unsigned u[8]; };

__device__ __forceinline__ v16bf frag_cast(const Frag& f) {
  return __builtin_bit_cast(v16bf, f);
}

__device__ __forceinline__ unsigned short f2bf(float x) {
  unsigned u = __builtin_bit_cast(unsigned, x);
  u += 0x7fffu + ((u >> 16) & 1u);            // round-to-nearest-even
  return (unsigned short)(u >> 16);
}

__device__ __forceinline__ v8f wmma_bf16(const Frag& a, const Frag& b, v8f c) {
  return __builtin_amdgcn_wmma_f32_16x16x32_bf16(
      false, frag_cast(a), false, frag_cast(b), (short)0, c, false, false);
}

// A operand: [M][K] row-major bf16; 16x32 tile at (row0,k0).
__device__ __forceinline__ Frag loadA16(const unsigned short* base, int row0, int k0, int ld) {
  int lane = threadIdx.x & 31;
  int m = lane & 15, half = lane >> 4;
  const unsigned short* rp = base + (row0 + m) * ld + k0;
  Frag f;
#pragma unroll
  for (int v = 0; v < 8; ++v) {
    int k = (half << 4) + (v << 1);
    f.u[v] = *(const unsigned*)(rp + k);
  }
  return f;
}

// B operand = W^T with W [N][K] row-major bf16: B[k][n] = W[n0+n][k0+k]
__device__ __forceinline__ Frag loadBW(const unsigned short* base, int n0, int k0, int ld) {
  int lane = threadIdx.x & 31;
  int n = lane & 15, half = lane >> 4;
  const unsigned short* rp = base + (n0 + n) * ld + k0 + (half << 4);
  Frag f;
#pragma unroll
  for (int v = 0; v < 8; ++v) f.u[v] = *(const unsigned*)(rp + (v << 1));
  return f;
}

// B operand stored directly [K][N] row-major (used for transfer @ h)
__device__ __forceinline__ Frag loadBD(const unsigned short* base, int k0, int n0, int ld) {
  int lane = threadIdx.x & 31;
  int n = lane & 15, half = lane >> 4;
  Frag f;
#pragma unroll
  for (int v = 0; v < 8; ++v) {
    int k = k0 + (half << 4) + (v << 1);
    unsigned lo = base[(k + 0) * ld + n0 + n];
    unsigned hi = base[(k + 1) * ld + n0 + n];
    f.u[v] = lo | (hi << 16);
  }
  return f;
}

__device__ __forceinline__ float sigmoidf_(float x) { return 1.0f / (1.0f + __expf(-x)); }

// ---------------------------------------------------------------- converts
__global__ void f32_to_bf16_k(const float* __restrict__ s, unsigned short* __restrict__ d, int n) {
  int i = blockIdx.x * blockDim.x + threadIdx.x;
  if (i < n) d[i] = f2bf(s[i]);
}

// convert f32 [N][K] -> bf16 [N][ldp], zeroing pad columns k in [K, ldp)
__global__ void conv_pad_k(const float* __restrict__ s, unsigned short* __restrict__ d,
                           int N, int K, int ldp) {
  size_t i = (size_t)blockIdx.x * 256 + threadIdx.x;
  size_t tot = (size_t)N * ldp;
  if (i < tot) {
    int n = (int)(i / ldp), k = (int)(i % ldp);
    d[i] = (k < K) ? f2bf(s[(size_t)n * K + k]) : (unsigned short)0;
  }
}

// ---------------------------------------------------------------- scan
struct ScanArgs {
  const float *x, *emb_w, *emb_b;
  const unsigned short *wih, *whh;          // bf16 [384][64], [384][128]
  const float *bih, *bhh;
  const unsigned short *qw1, *qw2, *qw3, *kw1, *kw2, *kw3;  // bf16 [128][128]
  const float *qb1, *qb2, *qb3, *kb1, *kb2, *kb3;
  const float *gate_bias, *pred_w, *pred_b;
  float *mix_out;             // [B,T,C,C] region of d_out (f32)
  unsigned short *mix_bf;     // [M_PAD][K1P] bf16 copy in ws (GEMM1 A operand)
  float *pred_out;            // [B,T-1,C] region of d_out
};

__device__ void mlp_layer(const unsigned short* in, const unsigned short* w,
                          const float* bias, unsigned short* out, int relu) {
  int tid = threadIdx.x, wid = tid >> 5, lane = tid & 31;
  int half = lane >> 4, ln = lane & 15;
  for (int task = wid; task < MT7 * 8; task += 16) {
    int row0 = (task >> 3) << 4, j = (task & 7) << 4;
    v8f acc = {0.f,0.f,0.f,0.f,0.f,0.f,0.f,0.f};
#pragma unroll
    for (int k0 = 0; k0 < HH; k0 += 32)
      acc = wmma_bf16(loadA16(in, row0, k0, HH), loadBW(w, j, k0, HH), acc);
    int cn = j + ln;
    float bv = bias[cn];
#pragma unroll
    for (int v = 0; v < 8; ++v) {
      int m = row0 + v + (half << 3);
      float val = acc[v] + bv;
      if (relu) val = fmaxf(val, 0.f);
      out[m * HH + cn] = (m < CC) ? f2bf(val) : (unsigned short)0;
    }
  }
}

__global__ __launch_bounds__(512) void scan_kernel(ScanArgs A) {
  extern __shared__ unsigned char smem[];
  float*          h32   = (float*)         (smem + 0);       // 112*128 f32  = 57344
  unsigned short* hbf   = (unsigned short*)(smem + 57344);   // 28672
  unsigned short* embbf = (unsigned short*)(smem + 86016);   // 14336
  unsigned short* m1    = (unsigned short*)(smem + 100352);  // 28672
  unsigned short* m2    = (unsigned short*)(smem + 129024);  // 28672
  unsigned short* qbf   = (unsigned short*)(smem + 157696);  // 28672
  unsigned short* kbf   = (unsigned short*)(smem + 186368);  // 28672
  unsigned short* trbf  = (unsigned short*)(smem + 215040);  // 28672
  float*          scr   = (float*)         (smem + 243712);  // 57344 staging / tr32[112][112]
  float*          red   = (float*)         (smem + 301056);  // 16
  // total 301072 bytes (<= 320KB WGP LDS)

  int b = blockIdx.x;
  int tid = threadIdx.x, wid = tid >> 5, lane = tid & 31;
  int half = lane >> 4, ln = lane & 15;

  for (int i = tid; i < CPAD * HH; i += 512) { h32[i] = 0.f; hbf[i] = 0; }
  __syncthreads();

  for (int t = 0; t < TT; ++t) {
    // ---- embed
    for (int i = tid; i < CPAD * EE; i += 512) {
      int c = i >> 6, e = i & 63;
      float v = 0.f;
      if (c < CC) v = A.x[(b * TT + t) * CC + c] * A.emb_w[e] + A.emb_b[e];
      embbf[i] = f2bf(v);
    }
    __syncthreads();

    // ---- GRU gates via WMMA
    for (int task = wid; task < MT7 * 8; task += 16) {
      int row0 = (task >> 3) << 4, j = (task & 7) << 4;
      v8f ar = {0.f,0.f,0.f,0.f,0.f,0.f,0.f,0.f};
      v8f az = ar, agin = ar, aghn = ar;
#pragma unroll
      for (int k0 = 0; k0 < EE; k0 += 32) {
        Frag a = loadA16(embbf, row0, k0, EE);
        ar   = wmma_bf16(a, loadBW(A.wih, j,          k0, EE), ar);
        az   = wmma_bf16(a, loadBW(A.wih, HH + j,     k0, EE), az);
        agin = wmma_bf16(a, loadBW(A.wih, 2 * HH + j, k0, EE), agin);
      }
#pragma unroll
      for (int k0 = 0; k0 < HH; k0 += 32) {
        Frag a = loadA16(hbf, row0, k0, HH);
        ar   = wmma_bf16(a, loadBW(A.whh, j,          k0, HH), ar);
        az   = wmma_bf16(a, loadBW(A.whh, HH + j,     k0, HH), az);
        aghn = wmma_bf16(a, loadBW(A.whh, 2 * HH + j, k0, HH), aghn);
      }
      int cn = j + ln;
      float bir = A.bih[cn],           bhr = A.bhh[cn];
      float biz = A.bih[HH + cn],      bhz = A.bhh[HH + cn];
      float bin_ = A.bih[2 * HH + cn], bhn = A.bhh[2 * HH + cn];
#pragma unroll
      for (int v = 0; v < 8; ++v) {
        int m = row0 + v + (half << 3);
        float r  = sigmoidf_(ar[v] + bir + bhr);
        float z  = sigmoidf_(az[v] + biz + bhz);
        float nn = tanhf(agin[v] + bin_ + r * (aghn[v] + bhn));
        float hv = (1.f - z) * nn + z * h32[m * HH + cn];
        scr[m * HH + cn] = (m < CC) ? hv : 0.f;
      }
    }
    __syncthreads();
    for (int i = tid; i < CPAD * HH; i += 512) { float v = scr[i]; h32[i] = v; hbf[i] = f2bf(v); }
    __syncthreads();

    // ---- q / k MLPs
    mlp_layer(hbf, A.qw1, A.qb1, m1, 1);  __syncthreads();
    mlp_layer(m1,  A.qw2, A.qb2, m2, 1);  __syncthreads();
    mlp_layer(m2,  A.qw3, A.qb3, qbf, 0); __syncthreads();
    mlp_layer(hbf, A.kw1, A.kb1, m1, 1);  __syncthreads();
    mlp_layer(m1,  A.kw2, A.kb2, m2, 1);  __syncthreads();
    mlp_layer(m2,  A.kw3, A.kb3, kbf, 0); __syncthreads();

    // ---- transfer = q @ k^T  -> scr as tr32[112][112]
    for (int task = wid; task < MT7 * MT7; task += 16) {
      int i0 = (task / MT7) << 4, j0 = (task % MT7) << 4;
      v8f acc = {0.f,0.f,0.f,0.f,0.f,0.f,0.f,0.f};
#pragma unroll
      for (int k0 = 0; k0 < HH; k0 += 32)
        acc = wmma_bf16(loadA16(qbf, i0, k0, HH), loadBW(kbf, j0, k0, HH), acc);
#pragma unroll
      for (int v = 0; v < 8; ++v) {
        int ii = i0 + v + (half << 3), jj = j0 + ln;
        scr[ii * CPAD + jj] = (ii < CC && jj < CC) ? acc[v] : 0.f;
      }
    }
    if (tid == 0) red[0] = 0.f;
    __syncthreads();
    {
      float loc = 0.f;
      for (int i = tid; i < CPAD * CPAD; i += 512) { float v = scr[i]; loc += v * v; }
      atomicAdd(red, loc);
    }
    __syncthreads();
    float rnorm = 1.0f / sqrtf(red[0]);

    // ---- normalize, gate, emit mixing (f32 to d_out, bf16 to ws), build transfer A
    float* mixp = A.mix_out + (size_t)(b * TT + t) * (CC * CC);
    unsigned short* mbp = A.mix_bf + (size_t)(b * TT + t) * K1P;
    for (int i = tid; i < CPAD * HH; i += 512) {
      int ii = i >> 7, jj = i & 127;
      unsigned short o = 0;
      if (ii < CC && jj < CC) {
        float v = scr[ii * CPAD + jj] * rnorm;
        float g = sigmoidf_(fabsf(v) + A.gate_bias[ii * CC + jj]);
        float ov = v * g;
        mixp[ii * CC + jj] = ov;
        o = f2bf(ov);
        mbp[ii * CC + jj] = o;
      }
      trbf[i] = o;
    }
    __syncthreads();

    // ---- h = transfer @ h
    for (int task = wid; task < MT7 * 8; task += 16) {
      int i0 = (task >> 3) << 4, j0 = (task & 7) << 4;
      v8f acc = {0.f,0.f,0.f,0.f,0.f,0.f,0.f,0.f};
#pragma unroll
      for (int k0 = 0; k0 < HH; k0 += 32)
        acc = wmma_bf16(loadA16(trbf, i0, k0, HH), loadBD(hbf, k0, j0, HH), acc);
#pragma unroll
      for (int v = 0; v < 8; ++v) {
        int m = i0 + v + (half << 3), cn = j0 + ln;
        scr[m * HH + cn] = (m < CC) ? acc[v] : 0.f;
      }
    }
    __syncthreads();
    for (int i = tid; i < CPAD * HH; i += 512) { float v = scr[i]; h32[i] = v; hbf[i] = f2bf(v); }
    if (t < TT - 1) {
      for (int c = tid; c < CC; c += 512) {
        float s = A.pred_b[0];
        for (int e = 0; e < HH; ++e) s += scr[c * HH + e] * A.pred_w[e];
        A.pred_out[(size_t)(b * (TT - 1) + t) * CC + c] = s;
      }
    }
    __syncthreads();
  }
}

// --------------------------------------------- async staging (ASYNCcnt path)
// copies 128 rows x 32 cols bf16 (8KB) as 512 x b128 chunks, 256 threads -> 2 each
__device__ __forceinline__ void stage_async_8k(const unsigned short* g, int ld, unsigned lds) {
  int tid = threadIdx.x;
#pragma unroll
  for (int c = 0; c < 2; ++c) {
    int idx = tid + (c << 8);
    int r = idx >> 2, col = (idx & 3) << 3;
    unsigned long long src = (unsigned long long)(uintptr_t)(g + (size_t)r * ld + col);
    unsigned dst = lds + (unsigned)(((r << 5) + col) << 1);
    asm volatile("global_load_async_to_lds_b128 %0, %1, off"
                 :: "v"(dst), "v"(src) : "memory");
  }
}

__device__ __forceinline__ void wait_async0() {
  asm volatile("s_wait_asynccnt 0x0" ::: "memory");
}

// --------------------------------------------- big classifier GEMM, bf16 operands,
// async double-buffered LDS staging. C = relu(A @ W^T + bias); store f32 or bf16.
__global__ __launch_bounds__(256) void gemm_bf16_async(
    const unsigned short* __restrict__ Ag, int lda,
    const unsigned short* __restrict__ Wg, int ldw,
    const float* __restrict__ bias,
    float* __restrict__ Cf, unsigned short* __restrict__ Cbf, int ldc,
    int M, int N, int Kp, int store_bf16) {
  __shared__ __align__(16) unsigned short Asb[2][128 * 32];
  __shared__ __align__(16) unsigned short Wsb[2][128 * 32];
  int tid = threadIdx.x;
  int w = tid >> 5, wm = w >> 2, wn = w & 3;   // 8 waves = 2 x 4
  int m0 = blockIdx.y * 128, n0 = blockIdx.x * 128;
  unsigned ldsA[2] = { (unsigned)(uintptr_t)(void*)&Asb[0][0],
                       (unsigned)(uintptr_t)(void*)&Asb[1][0] };
  unsigned ldsW[2] = { (unsigned)(uintptr_t)(void*)&Wsb[0][0],
                       (unsigned)(uintptr_t)(void*)&Wsb[1][0] };
  v8f acc[4][2];
#pragma unroll
  for (int mi = 0; mi < 4; ++mi)
#pragma unroll
    for (int nj = 0; nj < 2; ++nj) acc[mi][nj] = (v8f){0.f,0.f,0.f,0.f,0.f,0.f,0.f,0.f};

  const unsigned short* Abase = Ag + (size_t)m0 * lda;
  const unsigned short* Wbase = Wg + (size_t)n0 * ldw;
  int nK = Kp >> 5;
  stage_async_8k(Abase, lda, ldsA[0]);
  stage_async_8k(Wbase, ldw, ldsW[0]);
  wait_async0();
  __syncthreads();

  int cur = 0;
  for (int kt = 0; kt < nK; ++kt) {
    if (kt + 1 < nK) {       // prefetch next tile into the other buffer (async)
      int k0 = (kt + 1) << 5;
      stage_async_8k(Abase + k0, lda, ldsA[cur ^ 1]);
      stage_async_8k(Wbase + k0, ldw, ldsW[cur ^ 1]);
    }
    const unsigned short* As = Asb[cur];
    const unsigned short* Ws = Wsb[cur];
    Frag a[4], bfr[2];
#pragma unroll
    for (int mi = 0; mi < 4; ++mi) a[mi] = loadA16(As, wm * 64 + mi * 16, 0, 32);
#pragma unroll
    for (int nj = 0; nj < 2; ++nj) bfr[nj] = loadBW(Ws, wn * 32 + nj * 16, 0, 32);
#pragma unroll
    for (int mi = 0; mi < 4; ++mi)
#pragma unroll
      for (int nj = 0; nj < 2; ++nj) acc[mi][nj] = wmma_bf16(a[mi], bfr[nj], acc[mi][nj]);
    wait_async0();
    __syncthreads();
    cur ^= 1;
  }

  int lane = tid & 31, half = lane >> 4, ln = lane & 15;
#pragma unroll
  for (int mi = 0; mi < 4; ++mi)
#pragma unroll
    for (int nj = 0; nj < 2; ++nj)
#pragma unroll
      for (int v = 0; v < 8; ++v) {
        int m = m0 + wm * 64 + mi * 16 + v + (half << 3);
        int n = n0 + wn * 32 + nj * 16 + ln;
        if (m < M && n < N) {
          float val = fmaxf(acc[mi][nj][v] + bias[n], 0.f);
          if (store_bf16) Cbf[(size_t)m * ldc + n] = f2bf(val);
          else            Cf[(size_t)m * ldc + n] = val;
        }
      }
}

// ---------------------------------------------------------------- tail kernels
__global__ __launch_bounds__(128) void final_linear_k(
    const float* __restrict__ hc2, const float* __restrict__ w3,
    const float* __restrict__ b3, float* __restrict__ tl) {
  int m = blockIdx.x, tid = threadIdx.x;
  __shared__ float s0[128], s1[128];
  float p0 = 0.f, p1 = 0.f;
  for (int k = tid; k < N2; k += 128) {
    float h = hc2[(size_t)m * N2 + k];
    p0 += h * w3[k];
    p1 += h * w3[N2 + k];
  }
  s0[tid] = p0; s1[tid] = p1;
  __syncthreads();
  for (int s = 64; s > 0; s >>= 1) {
    if (tid < s) { s0[tid] += s0[tid + s]; s1[tid] += s1[tid + s]; }
    __syncthreads();
  }
  if (tid == 0) { tl[m * 2 + 0] = s0[0] + b3[0]; tl[m * 2 + 1] = s1[0] + b3[1]; }
}

__global__ void mean_logits_k(const float* __restrict__ tl, float* __restrict__ lg) {
  int tid = threadIdx.x;
  if (tid < BB * 2) {
    int b = tid >> 1, o = tid & 1;
    float s = 0.f;
    for (int t = 0; t < TT; ++t) s += tl[(b * TT + t) * 2 + o];
    lg[b * 2 + o] = s / (float)TT;
  }
}

// ---------------------------------------------------------------- launch
extern "C" void kernel_launch(void* const* d_in, const int* in_sizes, int n_in,
                              void* d_out, int out_size, void* d_ws, size_t ws_size,
                              hipStream_t stream) {
  (void)in_sizes; (void)n_in; (void)out_size; (void)ws_size;
  unsigned char* ws = (unsigned char*)d_ws;
  // ---- ws layout (256B aligned)
  unsigned short* wih_bf = (unsigned short*)(ws + 0);           // 384*64
  unsigned short* whh_bf = (unsigned short*)(ws + 49152);       // 384*128
  unsigned short* qw1 = (unsigned short*)(ws + 147456);
  unsigned short* qw2 = (unsigned short*)(ws + 180224);
  unsigned short* qw3 = (unsigned short*)(ws + 212992);
  unsigned short* kw1 = (unsigned short*)(ws + 245760);
  unsigned short* kw2 = (unsigned short*)(ws + 278528);
  unsigned short* kw3 = (unsigned short*)(ws + 311296);
  unsigned short* mix_bf = (unsigned short*)(ws + 344064);      // [M_PAD][K1P]  48,717,824 B
  unsigned short* w1_bf  = (unsigned short*)(ws + 49061888);    // [N1P][K1P]   102,563,840 B
  unsigned short* w2_bf  = (unsigned short*)(ws + 151625728);   // [N2P][K2P]    25,722,880 B
  unsigned short* hc1_bf = (unsigned short*)(ws + 177348608);   // [M_PAD][K2P]  24,436,736 B
  float*          hc2    = (float*)         (ws + 201785344);   // [2400][2500]  24,000,000 B

  float* outf     = (float*)d_out;
  float* logits   = outf;                 // 32
  float* mix_out  = outf + 32;            // 24,000,000
  float* tl_out   = outf + 24000032;      // 4,800
  float* pred_out = outf + 24004832;      // 238,400

  auto cf = [&](int idx, unsigned short* dst, int n) {
    f32_to_bf16_k<<<(n + 255) / 256, 256, 0, stream>>>((const float*)d_in[idx], dst, n);
  };
  cf(3, wih_bf, 384 * 64);
  cf(4, whh_bf, 384 * 128);
  cf(7, qw1, 128 * 128);  cf(9, qw2, 128 * 128);  cf(11, qw3, 128 * 128);
  cf(13, kw1, 128 * 128); cf(15, kw2, 128 * 128); cf(17, kw3, 128 * 128);

  // classifier weights -> bf16 with zeroed K-pad columns (makes padded-K WMMA exact)
  {
    size_t tot1 = (size_t)N1 * K1P;
    conv_pad_k<<<(unsigned)((tot1 + 255) / 256), 256, 0, stream>>>(
        (const float*)d_in[22], w1_bf, N1, K1, K1P);
    size_t tot2 = (size_t)N2 * K2P;
    conv_pad_k<<<(unsigned)((tot2 + 255) / 256), 256, 0, stream>>>(
        (const float*)d_in[24], w2_bf, N2, 5000, K2P);
  }

  ScanArgs SA;
  SA.x = (const float*)d_in[0];  SA.emb_w = (const float*)d_in[1];  SA.emb_b = (const float*)d_in[2];
  SA.wih = wih_bf; SA.whh = whh_bf;
  SA.bih = (const float*)d_in[5]; SA.bhh = (const float*)d_in[6];
  SA.qw1 = qw1; SA.qw2 = qw2; SA.qw3 = qw3; SA.kw1 = kw1; SA.kw2 = kw2; SA.kw3 = kw3;
  SA.qb1 = (const float*)d_in[8];  SA.qb2 = (const float*)d_in[10]; SA.qb3 = (const float*)d_in[12];
  SA.kb1 = (const float*)d_in[14]; SA.kb2 = (const float*)d_in[16]; SA.kb3 = (const float*)d_in[18];
  SA.gate_bias = (const float*)d_in[19];
  SA.pred_w = (const float*)d_in[20]; SA.pred_b = (const float*)d_in[21];
  SA.mix_out = mix_out; SA.mix_bf = mix_bf; SA.pred_out = pred_out;

  const int SMEM = 301072;
  hipFuncSetAttribute((const void*)scan_kernel,
                      hipFuncAttributeMaxDynamicSharedMemorySize, SMEM);
  scan_kernel<<<BB, 512, SMEM, stream>>>(SA);

  // GEMM1: hc1_bf = relu(mix_bf @ w1^T + b1)   [2400 x 5000], K=10016(padded)
  dim3 g1(N1P / 128, M_PAD / 128);
  gemm_bf16_async<<<g1, 256, 0, stream>>>(mix_bf, K1P, w1_bf, K1P,
                                          (const float*)d_in[23],
                                          nullptr, hc1_bf, K2P,
                                          M_CLF, N1, K1P, 1);
  // GEMM2: hc2 = relu(hc1_bf @ w2^T + b2)      [2400 x 2500], K=5024(padded)
  dim3 g2(N2P / 128, M_PAD / 128);
  gemm_bf16_async<<<g2, 256, 0, stream>>>(hc1_bf, K2P, w2_bf, K2P,
                                          (const float*)d_in[25],
                                          hc2, nullptr, N2,
                                          M_CLF, N2, K2P, 0);

  final_linear_k<<<M_CLF, 128, 0, stream>>>(hc2, (const float*)d_in[26],
                                            (const float*)d_in[27], tl_out);
  mean_logits_k<<<1, 64, 0, stream>>>(tl_out, logits);
}
